// NeRFNetwork_21930103013788
// MI455X (gfx1250) — matrix operand order repaired and checked by reading, not compiled, not run on hardware
//
#include <hip/hip_runtime.h>

typedef __attribute__((ext_vector_type(16))) _Float16 v16h;
typedef __attribute__((ext_vector_type(8)))  _Float16 v8h;
typedef __attribute__((ext_vector_type(8)))  float    v8f;

#define N_LEVELS   20
#define LOG2_T     24
#define TABLE_MASK ((1u << LOG2_T) - 1u)
// levels >= this are fully hashed (res^3 > 2^24): random, L2-thrashing -> NT
#define NT_LEVEL   9

// res-1 per level: floor(16 * 1.3819^l) - 1
__constant__ float c_resm1[N_LEVELS] = {
    15.f, 21.f, 29.f, 41.f, 57.f, 79.f, 110.f, 152.f, 211.f, 293.f,
    405.f, 560.f, 774.f, 1071.f, 1480.f, 2046.f, 2828.f, 3909.f, 5402.f, 7466.f};

static __device__ inline v8f wmma16(v16h a, v16h b, v8f c) {
    return __builtin_amdgcn_wmma_f32_16x16x32_f16(
        /*neg_a=*/false, a, /*neg_b=*/false, b,
        /*c_mod=*/(short)0, c, /*reuse_a=*/false, /*reuse_b=*/false);
}

// Load 16x32 f16 A-operand tile from row-major LDS buffer of width W halves.
// ISA layout: lanes 0-15 hold row M=lane, K = 0..7 then 16..23;
//             lanes 16-31 hold row M=lane-16, K = 8..15 then 24..31.
static __device__ inline v16h load_a(const _Float16* base, int W, int kofs, int lane) {
    int row = lane & 15;
    int kb  = (lane >> 4) * 8;
    const _Float16* p = base + row * W + kofs + kb;
    v8h lo = *(const v8h*)p;
    v8h hi = *(const v8h*)(p + 16);
    return __builtin_shufflevector(lo, hi, 0, 1, 2, 3, 4, 5, 6, 7,
                                   8, 9, 10, 11, 12, 13, 14, 15);
}

// Pre-packed B tile: 32 lanes x 16 contiguous halves.
static __device__ inline v16h load_b(const _Float16* base, int tile, int lane) {
    return *(const v16h*)(base + (size_t)tile * 512 + (size_t)lane * 16);
}

// Scatter one 16x16 f32 C tile (lane=column, VGPR v -> row) into row-major
// f16 LDS buffer of width W at column offset nofs, with optional ReLU.
static __device__ inline void store_c16(_Float16* base, int W, int nofs, int lane,
                                        v8f a, bool relu) {
    int n  = lane & 15;
    int mo = (lane >> 4) * 8;
#pragma unroll
    for (int v = 0; v < 8; ++v) {
        float f = a[v];
        if (relu) f = f > 0.f ? f : 0.f;
        base[(mo + v) * W + nofs + n] = (_Float16)f;
    }
}

// ---- prep: convert f32 weights to f16 in WMMA B-operand layout ------------
// B 32x16 f16 layout: lane = column N (lane&15); lanes 0-15 hold K=kt*32+0..7 then 16..23,
// lanes 16-31 hold K=kt*32+8..15 then 24..31, in the lane's 16 halves.
__global__ void pack_b(const float* __restrict__ W, int K, int Ncols, int NT,
                       _Float16* __restrict__ dst) {
    int tile = blockIdx.x;
    int kt = tile / NT, nt = tile % NT;
    int l   = threadIdx.x;
    int col = nt * 16 + (l & 15);
    int kb  = kt * 32 + (l >> 4) * 8;
    _Float16 vals[16];
#pragma unroll
    for (int j = 0; j < 16; ++j) {
        int   k = kb + (j < 8 ? j : j + 8);
        float v = (k < K && col < Ncols) ? W[k * Ncols + col] : 0.f;
        vals[j] = (_Float16)v;
    }
    *(v16h*)(dst + (size_t)tile * 512 + (size_t)l * 16) = *(v16h*)vals;
}

// ---- fused hash-encode + MLP ----------------------------------------------
__global__ __launch_bounds__(256) void nerf_fused(
    const float* __restrict__ x, const unsigned* __restrict__ tbl,
    const _Float16* __restrict__ wp, float* __restrict__ out, int Npts) {
    __shared__ _Float16 s_e[8][16 * 64];  // per-wave activation tile (K padded)
    __shared__ _Float16 s_g[8][16 * 32];  // per-wave g tile (15 -> pad 32)

    const int lane = threadIdx.x & 31;
    const int wave = threadIdx.x >> 5;
    const int row  = lane & 15;   // point within wave tile / WMMA column id
    const int grp  = lane >> 4;   // which half of the levels this lane does

    _Float16* eb = s_e[wave];
    _Float16* gb = s_g[wave];

    __builtin_prefetch((const void*)wp, 0, 0);  // pull packed weights toward us

    // zero-init pads (wave-private; block runs exactly once over its tile)
    {
        v16h z = {};
        *(v16h*)(eb + lane * 32)      = z;
        *(v16h*)(eb + lane * 32 + 16) = z;
        *(v16h*)(gb + lane * 16)      = z;
    }

    // ---- hash-grid encode: 2 lanes per point, 10 levels each --------------
    int p = blockIdx.x * 128 + wave * 16 + row;
    if (p >= Npts) p = Npts - 1;  // keep EXEC full for WMMA
    const float px = x[3 * p + 0], py = x[3 * p + 1], pz = x[3 * p + 2];
    const int lev0 = grp * 10;
    for (int i = 0; i < 10; ++i) {
        int   lev = lev0 + i;
        float rm1 = c_resm1[lev];
        float fx = px * rm1, fy = py * rm1, fz = pz * rm1;
        float gx = floorf(fx), gy = floorf(fy), gz = floorf(fz);
        float wx1 = fx - gx, wy1 = fy - gy, wz1 = fz - gz;
        float wx0 = 1.f - wx1, wy0 = 1.f - wy1, wz0 = 1.f - wz1;
        unsigned ix = (unsigned)(int)gx, iy = (unsigned)(int)gy, iz = (unsigned)(int)gz;
        unsigned hx0 = ix,               hx1 = ix + 1u;
        unsigned hy0 = iy * 2654435761u, hy1 = (iy + 1u) * 2654435761u;
        unsigned hz0 = iz * 805459861u,  hz1 = (iz + 1u) * 805459861u;
        const unsigned* t = tbl + ((unsigned long long)lev << LOG2_T);
        const bool nt = (lev >= NT_LEVEL);  // hashed levels: don't thrash L2
        float a0 = 0.f, a1 = 0.f;
#pragma unroll
        for (int c = 0; c < 8; ++c) {
            int ci = (c >> 2) & 1, cj = (c >> 1) & 1, ck = c & 1;
            unsigned h   = (ci ? hx1 : hx0) ^ (cj ? hy1 : hy0) ^ (ck ? hz1 : hz0);
            const unsigned* q32 = &t[h & TABLE_MASK];
            unsigned d;
            if (nt) d = __builtin_nontemporal_load(q32);  // th:NT — random 704MB set
            else    d = *q32;                             // RT — ~61MB set stays in L2
            float w = (ci ? wx1 : wx0) * (cj ? wy1 : wy0) * (ck ? wz1 : wz0);
            union { unsigned u; _Float16 f[2]; } q; q.u = d;
            a0 += (float)q.f[0] * w;
            a1 += (float)q.f[1] * w;
        }
        union { unsigned u; _Float16 f[2]; } o;
        o.f[0] = (_Float16)a0; o.f[1] = (_Float16)a1;
        *(unsigned*)(eb + row * 64 + lev * 2) = o.u;  // e[row][2*lev..2*lev+1]
    }
    asm volatile("s_wait_dscnt 0" ::: "memory");  // wave-private LDS handoff

    // packed-weight table offsets (halves)
    const _Float16* wWs0 = wp;           // 2x4 tiles (K=40->64, N=64)
    const _Float16* wWs1 = wp + 4096;    // 2x1 tiles (K=64,     N=16)
    const _Float16* wWc0 = wp + 5120;    // 1x4 tiles (K=15->32, N=64)
    const _Float16* wWc1 = wp + 7168;    // 2x4 tiles (K=64,     N=64)
    const _Float16* wWc2 = wp + 11264;   // 2x1 tiles (K=64,     N=3->16)

    // ---- layer 0: h = relu(e @ Ws0)  [16x64] ------------------------------
    v16h a0 = load_a(eb, 64, 0, lane);
    v16h a1 = load_a(eb, 64, 32, lane);
    v8f  acc[4];
#pragma unroll
    for (int nt = 0; nt < 4; ++nt) {
        v8f c = {};
        c = wmma16(a0, load_b(wWs0, nt, lane), c);
        c = wmma16(a1, load_b(wWs0, 4 + nt, lane), c);
        acc[nt] = c;
    }
#pragma unroll
    for (int nt = 0; nt < 4; ++nt) store_c16(eb, 64, nt * 16, lane, acc[nt], true);
    asm volatile("s_wait_dscnt 0" ::: "memory");

    // ---- s_o = h @ Ws1  [16x16] -------------------------------------------
    a0 = load_a(eb, 64, 0, lane);
    a1 = load_a(eb, 64, 32, lane);
    v8f so = {};
    so = wmma16(a0, load_b(wWs1, 0, lane), so);
    so = wmma16(a1, load_b(wWs1, 1, lane), so);

    // scatter: col 0 -> sigma output, cols 1..15 -> g tile (A layout source)
    const int mo   = grp * 8;
    float*    outp = out + (size_t)(blockIdx.x * 128 + wave * 16) * 4;
#pragma unroll
    for (int v = 0; v < 8; ++v) {
        float val = so[v];
        if (row == 0) __builtin_nontemporal_store(val, &outp[(mo + v) * 4 + 0]);
        else          gb[(mo + v) * 32 + (row - 1)] = (_Float16)val;
    }
    asm volatile("s_wait_dscnt 0" ::: "memory");

    // ---- c1 = relu(g @ Wc0)  [16x64] --------------------------------------
    v16h ga = load_a(gb, 32, 0, lane);
#pragma unroll
    for (int nt = 0; nt < 4; ++nt) {
        v8f c = {};
        acc[nt] = wmma16(ga, load_b(wWc0, nt, lane), c);
    }
#pragma unroll
    for (int nt = 0; nt < 4; ++nt) store_c16(eb, 64, nt * 16, lane, acc[nt], true);
    asm volatile("s_wait_dscnt 0" ::: "memory");

    // ---- c2 = relu(c1 @ Wc1)  [16x64] -------------------------------------
    a0 = load_a(eb, 64, 0, lane);
    a1 = load_a(eb, 64, 32, lane);
#pragma unroll
    for (int nt = 0; nt < 4; ++nt) {
        v8f c = {};
        c = wmma16(a0, load_b(wWc1, nt, lane), c);
        c = wmma16(a1, load_b(wWc1, 4 + nt, lane), c);
        acc[nt] = c;
    }
#pragma unroll
    for (int nt = 0; nt < 4; ++nt) store_c16(eb, 64, nt * 16, lane, acc[nt], true);
    asm volatile("s_wait_dscnt 0" ::: "memory");

    // ---- c3 = c2 @ Wc2  [16x16], cols 0..2 valid --------------------------
    a0 = load_a(eb, 64, 0, lane);
    a1 = load_a(eb, 64, 32, lane);
    v8f co = {};
    co = wmma16(a0, load_b(wWc2, 0, lane), co);
    co = wmma16(a1, load_b(wWc2, 1, lane), co);
#pragma unroll
    for (int v = 0; v < 8; ++v) {
        if (row < 3) __builtin_nontemporal_store(co[v], &outp[(mo + v) * 4 + 1 + row]);
    }
}

extern "C" void kernel_launch(void* const* d_in, const int* in_sizes, int n_in,
                              void* d_out, int out_size, void* d_ws, size_t ws_size,
                              hipStream_t stream) {
    const float*    x   = (const float*)d_in[0];
    const unsigned* tbl = (const unsigned*)d_in[1];  // fp16 pairs as u32
    const float*    Ws0 = (const float*)d_in[2];     // 40x64
    const float*    Ws1 = (const float*)d_in[3];     // 64x16
    const float*    Wc0 = (const float*)d_in[4];     // 15x64
    const float*    Wc1 = (const float*)d_in[5];     // 64x64
    const float*    Wc2 = (const float*)d_in[6];     // 64x3
    float*          out = (float*)d_out;
    int             Npts = in_sizes[0] / 3;

    _Float16* wp = (_Float16*)d_ws;  // 12288 halves = 24 KB of packed weights

    pack_b<<<8, 32, 0, stream>>>(Ws0, 40, 64, 4, wp + 0);
    pack_b<<<2, 32, 0, stream>>>(Ws1, 64, 16, 1, wp + 4096);
    pack_b<<<4, 32, 0, stream>>>(Wc0, 15, 64, 4, wp + 5120);
    pack_b<<<8, 32, 0, stream>>>(Wc1, 64, 64, 4, wp + 7168);
    pack_b<<<2, 32, 0, stream>>>(Wc2, 64, 3, 1, wp + 11264);

    int blocks = (Npts + 127) / 128;  // 8 waves x 16 points per block
    nerf_fused<<<blocks, 256, 0, stream>>>(x, tbl, wp, out, Npts);
}
